// GNNEncoder_15848429322408
// MI455X (gfx1250) — matrix operand order repaired and checked by reading, not compiled, not run on hardware
//
#include <hip/hip_runtime.h>
#include <hip/hip_bf16.h>
#include <float.h>

// ---------------------------------------------------------------------------
// GNN PointNet encoder for MI455X (gfx1250, wave32, WMMA)
// B=64 graphs, P=1024 points, C=64 channels, OUT=6
// ---------------------------------------------------------------------------

#define GB 64      // graphs
#define GP 1024    // points per graph
#define GC 64      // hidden channels
#define GOUT 6

typedef __attribute__((ext_vector_type(16))) _Float16 v16h;
typedef __attribute__((ext_vector_type(8)))  float    v8f;

// ---------------------------------------------------------------------------
// Kernel 1: brute-force kNN (top-6, self included, sorted ascending distance).
// One block per graph. pos tile cached in LDS; register insertion sort.
// ---------------------------------------------------------------------------
__global__ __launch_bounds__(256) void knn_kernel(const float* __restrict__ pos,
                                                  int* __restrict__ knn) {
  const int b = blockIdx.x;
  __shared__ float sp[GP][3];                       // 12 KB
  for (int i = threadIdx.x; i < GP; i += 256) {
    sp[i][0] = pos[(size_t)(b * GP + i) * 3 + 0];
    sp[i][1] = pos[(size_t)(b * GP + i) * 3 + 1];
    sp[i][2] = pos[(size_t)(b * GP + i) * 3 + 2];
  }
  __syncthreads();
  for (int pi = threadIdx.x; pi < GP; pi += 256) {
    const float px = sp[pi][0], py = sp[pi][1], pz = sp[pi][2];
    float bd[6]; int bi[6];
#pragma unroll
    for (int t = 0; t < 6; ++t) { bd[t] = FLT_MAX; bi[t] = 0; }
    for (int j = 0; j < GP; ++j) {
      const float dx = sp[j][0] - px, dy = sp[j][1] - py, dz = sp[j][2] - pz;
      const float d2 = dx * dx + dy * dy + dz * dz;
      if (d2 < bd[5]) {
        bd[5] = d2; bi[5] = j;
#pragma unroll
        for (int t = 5; t > 0; --t) {
          if (bd[t] < bd[t - 1]) {
            float td = bd[t]; bd[t] = bd[t - 1]; bd[t - 1] = td;
            int   ti = bi[t]; bi[t] = bi[t - 1]; bi[t - 1] = ti;
          }
        }
      }
    }
#pragma unroll
    for (int t = 0; t < 6; ++t) knn[(size_t)(b * GP + pi) * 6 + t] = bi[t];
  }
}

// ---------------------------------------------------------------------------
// Native CDNA5 float atomic max (no return): relu(max_k m) == max(0, max_k m)
// with zero-initialized destination.
// ---------------------------------------------------------------------------
__device__ __forceinline__ void atomic_max_f32(float* p, float v) {
  asm volatile("global_atomic_max_num_f32 %0, %1, off" : : "v"(p), "v"(v) : "memory");
}

// ---------------------------------------------------------------------------
// Kernel 2: fused edge-MLP PointNet layer.
//   feat(e=(i,j)) = [h_i | h_j | pos_j - pos_i]   (F = 2*CIN+3)
//   m = relu(feat @ W1 + b1) @ W2 + b2            (two WMMA f16 GEMMs)
//   hout[i] = atomic-max over the KNBR edges of each point (output pre-zeroed)
// One block = 64 points of one graph -> E = 64*KNBR edge rows (mult. of 16).
// 256 threads = 8 waves; wave w owns M-tiles w, w+8, ...
// ---------------------------------------------------------------------------
template <int CIN, int KNBR>
__global__ __launch_bounds__(256) void layer_kernel(
    const float* __restrict__ hin, const float* __restrict__ pos,
    const int* __restrict__ knn, const float* __restrict__ W1,
    const float* __restrict__ b1, const float* __restrict__ W2,
    const float* __restrict__ b2, float* __restrict__ hout) {
  constexpr int F    = 2 * CIN + 3;
  constexpr int KT1  = (F + 31) / 32;   // K-tiles of GEMM1
  constexpr int TP   = 64;              // points per block
  constexpr int E    = TP * KNBR;       // edge rows per block
  constexpr int NUMM = E / 16;          // 16-row M-tiles

  const int b  = blockIdx.x / (GP / TP);
  const int p0 = (blockIdx.x % (GP / TP)) * TP;
  const int tid   = threadIdx.x;
  const int w     = tid >> 5;           // wave id (8 waves)
  const int lane  = tid & 31;
  const int laneM = lane & 15;
  const int hiH   = lane >> 4;          // 0 for lanes 0-15, 1 for 16-31

  // B-fragment-swizzled weights in LDS (lanes 0-15 -> K {0..7,16..23},
  // lanes 16-31 -> K {8..15,24..31}; column N = ntile*16 + lane%16).
  __shared__ alignas(32) _Float16 w1f[KT1][4][32][16];
  __shared__ alignas(32) _Float16 w2f[2][4][32][16];
  __shared__ alignas(32) _Float16 a2s[8][16][GC];   // per-wave GEMM1->GEMM2 restage

  for (int idx = tid; idx < KT1 * 4 * 32 * 16; idx += 256) {
    const int hh = idx & 15, ln = (idx >> 4) & 31, nt = (idx >> 9) & 3, kt = idx >> 11;
    const int kk = kt * 32 + hh + ((hh < 8) ? 0 : 8) + 8 * (ln >> 4);
    const int nn = nt * 16 + (ln & 15);
    w1f[kt][nt][ln][hh] = (_Float16)(kk < F ? W1[(size_t)kk * GC + nn] : 0.0f);
  }
  for (int idx = tid; idx < 2 * 4 * 32 * 16; idx += 256) {
    const int hh = idx & 15, ln = (idx >> 4) & 31, nt = (idx >> 9) & 3, kt = idx >> 11;
    const int kk = kt * 32 + hh + ((hh < 8) ? 0 : 8) + 8 * (ln >> 4);
    const int nn = nt * 16 + (ln & 15);
    w2f[kt][nt][ln][hh] = (_Float16)W2[(size_t)kk * GC + nn];
  }
  __syncthreads();

  float bias1[4], bias2[4];
#pragma unroll
  for (int nt = 0; nt < 4; ++nt) {
    bias1[nt] = b1[nt * 16 + laneM];
    bias2[nt] = b2[nt * 16 + laneM];
  }

  for (int mt = w; mt < NUMM; mt += 8) {
    // This lane supplies A-matrix row M = lane%16 of the M-tile.
    const int e  = mt * 16 + laneM;
    const int pl = e / KNBR;                         // local point
    const int s  = e % KNBR;                         // neighbor slot
    const int gi = b * GP + p0 + pl;
    const int gj = b * GP + knn[(size_t)gi * 6 + s];

    // ---------------- GEMM1: [16 x F] x [F x 64], f16 in, f32 acc ----------
    v8f acc[4];
#pragma unroll
    for (int nt = 0; nt < 4; ++nt)
#pragma unroll
      for (int r = 0; r < 8; ++r) acc[nt][r] = bias1[nt];

#pragma unroll
    for (int kt = 0; kt < KT1; ++kt) {
      v16h a;
#pragma unroll
      for (int t = 0; t < 16; ++t) {
        // 16-bit A layout: lo lanes K in {0..7,16..23}, hi lanes {8..15,24..31}
        const int Kf = kt * 32 + t + ((t < 8) ? 0 : 8) + 8 * hiH;
        float fv;
        if (Kf < CIN)                fv = hin[(size_t)gi * CIN + Kf];
        else if (Kf < 2 * CIN)       fv = hin[(size_t)gj * CIN + (Kf - CIN)];
        else if (Kf < 2 * CIN + 3) { const int d = Kf - 2 * CIN;
                                     fv = pos[(size_t)gj * 3 + d] - pos[(size_t)gi * 3 + d]; }
        else                         fv = 0.0f;
        a[t] = (_Float16)fv;
      }
#pragma unroll
      for (int nt = 0; nt < 4; ++nt) {
        const v16h bf = *(const v16h*)&w1f[kt][nt][lane][0];
        acc[nt] = __builtin_amdgcn_wmma_f32_16x16x32_f16(false, a, false, bf,
                                                         (short)0, acc[nt], false, false);
      }
    }

    // ---------------- relu + restage C-layout -> A-layout via LDS ----------
#pragma unroll
    for (int nt = 0; nt < 4; ++nt)
#pragma unroll
      for (int r = 0; r < 8; ++r) {
        float v = acc[nt][r];
        v = v > 0.0f ? v : 0.0f;
        a2s[w][r + 8 * hiH][nt * 16 + laneM] = (_Float16)v;   // row M, col N
      }

    // ---------------- GEMM2: [16 x 64] x [64 x 64] -------------------------
    v8f acc2[4];
#pragma unroll
    for (int nt = 0; nt < 4; ++nt)
#pragma unroll
      for (int r = 0; r < 8; ++r) acc2[nt][r] = bias2[nt];

#pragma unroll
    for (int kt = 0; kt < 2; ++kt) {
      v16h a2;
#pragma unroll
      for (int t = 0; t < 16; ++t) {
        const int K = kt * 32 + t + ((t < 8) ? 0 : 8) + 8 * hiH;
        a2[t] = a2s[w][laneM][K];
      }
#pragma unroll
      for (int nt = 0; nt < 4; ++nt) {
        const v16h bf = *(const v16h*)&w2f[kt][nt][lane][0];
        acc2[nt] = __builtin_amdgcn_wmma_f32_16x16x32_f16(false, a2, false, bf,
                                                          (short)0, acc2[nt], false, false);
      }
    }

    // ------------- fused neighbor-max + outer relu via atomic max ----------
#pragma unroll
    for (int nt = 0; nt < 4; ++nt)
#pragma unroll
      for (int r = 0; r < 8; ++r) {
        const int M  = r + 8 * hiH;                  // edge row in this M-tile
        const int e2 = mt * 16 + M;
        const int pp = b * GP + p0 + e2 / KNBR;
        atomic_max_f32(&hout[(size_t)pp * GC + nt * 16 + laneM], acc2[nt][r]);
      }
  }
}

// ---------------------------------------------------------------------------
// Kernel 3: global max pool over P + 64->6 linear head. One block per graph.
// ---------------------------------------------------------------------------
__global__ __launch_bounds__(64) void pool_reg_kernel(const float* __restrict__ h,
                                                      const float* __restrict__ regW,
                                                      const float* __restrict__ regb,
                                                      float* __restrict__ out) {
  const int b = blockIdx.x, c = threadIdx.x;
  __shared__ float pooled[GC];
  float m = h[(size_t)(b * GP) * GC + c];
  for (int p = 1; p < GP; ++p)
    m = fmaxf(m, h[(size_t)(b * GP + p) * GC + c]);
  pooled[c] = m;
  __syncthreads();
  if (c < GOUT) {
    float s = regb[c];
#pragma unroll 8
    for (int cc = 0; cc < GC; ++cc) s += pooled[cc] * regW[cc * GOUT + c];
    out[b * GOUT + c] = s;
  }
}

// ---------------------------------------------------------------------------
// Host launcher (graph-capture safe: only kernels + hipMemsetAsync on stream)
// ---------------------------------------------------------------------------
extern "C" void kernel_launch(void* const* d_in, const int* in_sizes, int n_in,
                              void* d_out, int out_size, void* d_ws, size_t ws_size,
                              hipStream_t stream) {
  // setup_inputs order: x, pos, batch, {convL_W1, convL_b1, convL_W2, convL_b2} x3, reg_W, reg_b
  const float* pos  = (const float*)d_in[1];
  const float* c1W1 = (const float*)d_in[3];
  const float* c1b1 = (const float*)d_in[4];
  const float* c1W2 = (const float*)d_in[5];
  const float* c1b2 = (const float*)d_in[6];
  const float* c2W1 = (const float*)d_in[7];
  const float* c2b1 = (const float*)d_in[8];
  const float* c2W2 = (const float*)d_in[9];
  const float* c2b2 = (const float*)d_in[10];
  const float* c3W1 = (const float*)d_in[11];
  const float* c3b1 = (const float*)d_in[12];
  const float* c3W2 = (const float*)d_in[13];
  const float* c3b2 = (const float*)d_in[14];
  const float* regW = (const float*)d_in[15];
  const float* regb = (const float*)d_in[16];
  float* out = (float*)d_out;

  char* ws = (char*)d_ws;
  const size_t HBYTES = (size_t)GB * GP * GC * sizeof(float);   // 16 MB
  int*   knn = (int*)ws;
  float* hA  = (float*)(ws + (size_t)(2 << 20));
  float* hB  = (float*)(ws + (size_t)(2 << 20) + HBYTES);

  // 1) kNN (top-6 once; k=4 and k=3 are sorted prefixes)
  knn_kernel<<<GB, 256, 0, stream>>>(pos, knn);

  // 2) layer 1: h = pos (Cin=3), k=6  -> hA (zero-init feeds fused relu-max)
  hipMemsetAsync(hA, 0, HBYTES, stream);
  layer_kernel<3, 6><<<GB * (GP / 64), 256, 0, stream>>>(pos, pos, knn,
                                                         c1W1, c1b1, c1W2, c1b2, hA);
  // 3) layer 2: Cin=64, k=4 -> hB
  hipMemsetAsync(hB, 0, HBYTES, stream);
  layer_kernel<64, 4><<<GB * (GP / 64), 256, 0, stream>>>(hA, pos, knn,
                                                          c2W1, c2b1, c2W2, c2b2, hB);
  // 4) layer 3: Cin=64, k=3 -> hA (reused)
  hipMemsetAsync(hA, 0, HBYTES, stream);
  layer_kernel<64, 3><<<GB * (GP / 64), 256, 0, stream>>>(hB, pos, knn,
                                                          c3W1, c3b1, c3W2, c3b2, hA);
  // 5) pool + regression head
  pool_reg_kernel<<<GB, 64, 0, stream>>>(hA, regW, regb, out);
}
